// MultiHeadAttentionLayer_38285338476696
// MI455X (gfx1250) — matrix operand order repaired and checked by reading, not compiled, not run on hardware
//
#include <hip/hip_runtime.h>
#include <hip/hip_bf16.h>

// ---------------------------------------------------------------------------
// Graph multi-head attention layer, CDNA5 (gfx1250) wave32 + WMMA bf16.
// N=50000 nodes, E=800000 edges, IN_DIM=128, H=8 heads, D=16 (H*D = 128).
//
// GEMMs run on v_wmma_f32_16x16x32_bf16 (f32 accumulate). Weights are
// pre-packed once into per-lane B-fragment layout (bf16, 32B/lane) so the
// hot loops load fragments with single coalesced 32-byte vector loads.
// e/kr/e_out use non-temporal hints (streamed once); gather tables stay
// cache-resident. Kernels are templated on FULL so the fully-valid tiles
// (all of them, since 16 | N and 16 | E) carry no per-lane guards and no
// exec-mask save/restore cascades.
// ---------------------------------------------------------------------------

typedef __attribute__((ext_vector_type(16))) __bf16 v16bf;
typedef __attribute__((ext_vector_type(8)))  float  v8f;
typedef __attribute__((ext_vector_type(4)))  float  f32x4;

struct AFrag { v16bf f[4]; };   // 16 rows x 128 K, four K=32 blocks

__device__ __forceinline__ v8f wmma_bf16(v16bf a, v16bf b, v8f c) {
    // D = A(16x32 bf16) * B(32x16 bf16) + C(16x16 f32)
    return __builtin_amdgcn_wmma_f32_16x16x32_bf16(
        /*neg_a=*/false, a, /*neg_b=*/false, b,
        /*c_mod=*/(short)0, c, /*reuse_a=*/false, /*reuse_b=*/false);
}

// A fragment: lane holds row m (= lane&15).  Element j maps to
//   K = 32*kb + 8*half + j + (j>=8 ? 8 : 0)     (ISA 7.12.2, 16-bit A 16x32)
// Per lane that is two runs of 8 contiguous floats -> 4x f32x4 loads (16B aligned).
template <bool NT>
__device__ __forceinline__ v16bf load_a_frag(const float* __restrict__ row,
                                             int kb, int half) {
    const float* p0 = row + kb * 32 + 8 * half;        // K run  [base .. base+7]
    const float* p1 = p0 + 16;                         // K run  [base+16 .. base+23]
    f32x4 r0, r1, r2, r3;
    if (NT) {
        r0 = __builtin_nontemporal_load((const f32x4*)p0);
        r1 = __builtin_nontemporal_load((const f32x4*)(p0 + 4));
        r2 = __builtin_nontemporal_load((const f32x4*)p1);
        r3 = __builtin_nontemporal_load((const f32x4*)(p1 + 4));
    } else {
        r0 = *(const f32x4*)p0;
        r1 = *(const f32x4*)(p0 + 4);
        r2 = *(const f32x4*)p1;
        r3 = *(const f32x4*)(p1 + 4);
    }
    v16bf a;
#pragma unroll
    for (int t = 0; t < 4; ++t) { a[t]      = (__bf16)r0[t]; }
#pragma unroll
    for (int t = 0; t < 4; ++t) { a[4 + t]  = (__bf16)r1[t]; }
#pragma unroll
    for (int t = 0; t < 4; ++t) { a[8 + t]  = (__bf16)r2[t]; }
#pragma unroll
    for (int t = 0; t < 4; ++t) { a[12 + t] = (__bf16)r3[t]; }
    return a;
}

// Pre-packed B fragment: one contiguous 32-byte v16bf per lane.
__device__ __forceinline__ v16bf load_b_frag_pk(const __bf16* __restrict__ Wp,
                                                int kb, int nt, int lane) {
    return *(const v16bf*)(Wp + ((((nt * 4 + kb) * 32) + lane) << 4));
}

// ---------------------------------------------------------------------------
// Kernel P: pack 5 weight matrices [128,128] f32 -> bf16 B-fragment layout.
// out element index: ((nt*4 + kb)*32 + lane)*16 + j ;
// value = W[(kb*32 + 8*half + j + (j>=8 ? 8 : 0))*128 + nt*16 + n],  n=lane&15, half=lane>>4.
// ---------------------------------------------------------------------------
__global__ void pack_weights_kernel(const float* __restrict__ W0, const float* __restrict__ W1,
                                    const float* __restrict__ W2, const float* __restrict__ W3,
                                    const float* __restrict__ W4, __bf16* __restrict__ out) {
    int idx = blockIdx.x * blockDim.x + threadIdx.x;
    if (idx >= 5 * 16384) return;
    const int m = idx >> 14;
    const float* W = (m == 0) ? W0 : (m == 1) ? W1 : (m == 2) ? W2 : (m == 3) ? W3 : W4;
    const int t    = idx & 16383;
    const int j    = t & 15;
    const int lane = (t >> 4) & 31;
    const int kb   = (t >> 9) & 3;
    const int nt   = (t >> 11) & 7;
    const int n    = lane & 15;
    const int half = lane >> 4;
    const int k    = kb * 32 + 8 * half + j + (j >= 8 ? 8 : 0);
    const int col  = nt * 16 + n;
    out[idx] = (__bf16)W[k * 128 + col];
}

// ---------------------------------------------------------------------------
// Kernel 0: zero the scatter accumulators (wV, z).
// ---------------------------------------------------------------------------
__global__ void zero_kernel(float* __restrict__ p, size_t count) {
    size_t i = (size_t)blockIdx.x * blockDim.x + threadIdx.x;
    if (i < count) p[i] = 0.0f;
}

// ---------------------------------------------------------------------------
// Kernel 1: node projections  Qh/Kh/Vh = h @ {Q,K,V}w + bias   [N,128] each.
// One wave handles a 16-row tile; block = 4 waves = 64 rows.
// FULL=true : every row of the tile is valid (no guards at all).
// ---------------------------------------------------------------------------
template <bool FULL>
__global__ void node_proj_kernel(const float* __restrict__ h,
                                 const __bf16* __restrict__ Qp, const float* __restrict__ Qb,
                                 const __bf16* __restrict__ Kp, const float* __restrict__ Kb,
                                 const __bf16* __restrict__ Vp, const float* __restrict__ Vb,
                                 float* __restrict__ Qh, float* __restrict__ Kh,
                                 float* __restrict__ Vh, int block_base, int nrows) {
    const int lane = threadIdx.x & 31;
    const int wave = threadIdx.x >> 5;
    const int row0 = ((blockIdx.x + block_base) * 4 + wave) * 16;
    if (!FULL) {
        if (row0 >= nrows) return;             // wave-uniform: EXEC stays all-1s
    }
    const int n    = lane & 15;
    const int half = lane >> 4;

    int arow = row0 + n;                       // A row for this lane
    if (!FULL && arow >= nrows) arow = nrows - 1;   // clamp (loads only)
    const float* rowp = h + (size_t)arow * 128;

    AFrag A;
#pragma unroll
    for (int kb = 0; kb < 4; ++kb) A.f[kb] = load_a_frag<false>(rowp, kb, half);

#pragma unroll
    for (int w = 0; w < 3; ++w) {
        const __bf16* Wp   = (w == 0) ? Qp : (w == 1) ? Kp : Vp;
        const float*  bias = (w == 0) ? Qb : (w == 1) ? Kb : Vb;
        float*        O    = (w == 0) ? Qh : (w == 1) ? Kh : Vh;
        for (int nt = 0; nt < 8; ++nt) {
            const int col = nt * 16 + n;
            v8f acc = {};
#pragma unroll
            for (int kb = 0; kb < 4; ++kb)
                acc = wmma_bf16(A.f[kb], load_b_frag_pk(Wp, kb, nt, lane), acc);
            const float bv = bias[col];
#pragma unroll
            for (int v = 0; v < 8; ++v) {      // C element v -> row v + 8*half
                const int orow = row0 + v + 8 * half;
                if (FULL || orow < nrows)
                    O[(size_t)orow * 128 + col] = acc[v] + bv;
            }
        }
    }
}

// ---------------------------------------------------------------------------
// Kernel 2 (fused): per 16-edge wave tile:
//   proj_e = e@Ew+Eb, proj_kr = kr@Rw+Rb  (WMMA, kept in registers)
//   score  = (Kh[src]*Qh[dst]*0.25 + proj_kr) * proj_e   -> e_out (streaming NT)
//   s      = exp(clip(sum_D score))        (16-lane shuffle tree; N-tile == head)
//   atomic scatter: wV[dst] += Vh[src]*s,  z[dst,head] += s
// ---------------------------------------------------------------------------
template <bool FULL>
__global__ void edge_kernel(const float* __restrict__ e,  const float* __restrict__ kr,
                            const int*   __restrict__ src, const int* __restrict__ dst,
                            const __bf16* __restrict__ Ewp, const float* __restrict__ Ebias,
                            const __bf16* __restrict__ Rwp, const float* __restrict__ Rbias,
                            const float* __restrict__ Qh, const float* __restrict__ Kh,
                            const float* __restrict__ Vh,
                            float* __restrict__ e_out, float* __restrict__ wV,
                            float* __restrict__ z, int block_base, int nedges) {
    const int lane = threadIdx.x & 31;
    const int wave = threadIdx.x >> 5;
    const int row0 = ((blockIdx.x + block_base) * 4 + wave) * 16;
    if (!FULL) {
        if (row0 >= nedges) return;
    }
    const int n    = lane & 15;
    const int half = lane >> 4;

    int arow = row0 + n;
    if (!FULL && arow >= nedges) arow = nedges - 1;
    const float* ep = e  + (size_t)arow * 128;
    const float* kp = kr + (size_t)arow * 128;

    AFrag Ae, Ar;
#pragma unroll
    for (int kb = 0; kb < 4; ++kb) {
        Ae.f[kb] = load_a_frag<true>(ep, kb, half);
        Ar.f[kb] = load_a_frag<true>(kp, kb, half);
    }

    // Edge endpoints for the 8 output rows this lane owns (rows v + 8*half).
    int sidx[8], didx[8];
#pragma unroll
    for (int v = 0; v < 8; ++v) {
        int r = row0 + v + 8 * half;
        if (!FULL && r >= nedges) r = nedges - 1;
        sidx[v] = src[r];
        didx[v] = dst[r];
    }

    const float rsqD = 0.25f;                  // 1/sqrt(D), D=16

    for (int nt = 0; nt < 8; ++nt) {           // nt == head index (D=16 per head)
        const int col = nt * 16 + n;
        v8f acc_e = {}, acc_r = {};
#pragma unroll
        for (int kb = 0; kb < 4; ++kb) {
            acc_e = wmma_bf16(Ae.f[kb], load_b_frag_pk(Ewp, kb, nt, lane), acc_e);
            acc_r = wmma_bf16(Ar.f[kb], load_b_frag_pk(Rwp, kb, nt, lane), acc_r);
        }
        const float eb = Ebias[col];
        const float rb = Rbias[col];

#pragma unroll
        for (int v = 0; v < 8; ++v) {
            const int  erow  = row0 + v + 8 * half;
            const bool valid = FULL || (erow < nedges);
            const int  si = sidx[v], di = didx[v];
            const size_t so   = (size_t)si * 128 + col;
            const size_t dofs = (size_t)di * 128 + col;

            const float pe = acc_e[v] + eb;    // proj_e
            const float pr = acc_r[v] + rb;    // proj_kr
            const float ks = Kh[so];
            const float qd = Qh[dofs];
            const float vs = Vh[so];

            const float score = (ks * qd * rsqD + pr) * pe;
            if (valid)
                __builtin_nontemporal_store(score, &e_out[(size_t)erow * 128 + col]);

            // per-head sum over D=16 -> reduce across the 16-lane half
            float hs = score;
            hs += __shfl_xor(hs, 1, 32);
            hs += __shfl_xor(hs, 2, 32);
            hs += __shfl_xor(hs, 4, 32);
            hs += __shfl_xor(hs, 8, 32);
            hs = fminf(fmaxf(hs, -5.0f), 5.0f);
            const float s = expf(hs);

            if (valid) {
                atomicAdd(&wV[dofs], vs * s);
                if (n == 0) atomicAdd(&z[(size_t)di * 8 + nt], s);
            }
        }
    }
}

// ---------------------------------------------------------------------------
// Kernel 3: h_out = wV / (z + 1e-6)
// ---------------------------------------------------------------------------
__global__ void finalize_kernel(const float* __restrict__ wV,
                                const float* __restrict__ z,
                                float* __restrict__ h_out, int nrows) {
    const size_t total = (size_t)nrows * 128;
    size_t i = (size_t)blockIdx.x * blockDim.x + threadIdx.x;
    if (i >= total) return;
    const int node = (int)(i >> 7);
    const int head = (int)((i & 127) >> 4);
    h_out[i] = wV[i] / (z[(size_t)node * 8 + head] + 1e-6f);
}

// ---------------------------------------------------------------------------
extern "C" void kernel_launch(void* const* d_in, const int* in_sizes, int n_in,
                              void* d_out, int out_size, void* d_ws, size_t ws_size,
                              hipStream_t stream) {
    const float* h   = (const float*)d_in[0];
    const float* e   = (const float*)d_in[1];
    const float* kr  = (const float*)d_in[2];
    const int*   src = (const int*)  d_in[3];
    const int*   dst = (const int*)  d_in[4];
    const float* Qw  = (const float*)d_in[5];
    const float* Qb  = (const float*)d_in[6];
    const float* Kw  = (const float*)d_in[7];
    const float* Kb  = (const float*)d_in[8];
    const float* Vw  = (const float*)d_in[9];
    const float* Vb  = (const float*)d_in[10];
    const float* Ew  = (const float*)d_in[11];
    const float* Eb  = (const float*)d_in[12];
    const float* Rw  = (const float*)d_in[13];
    const float* Rb  = (const float*)d_in[14];

    const int N = in_sizes[0] / 128;   // 50000
    const int E = in_sizes[3];         // 800000

    // Workspace layout (floats): Qh | Kh | Vh | wV | z | packed weights (bf16)
    float* ws = (float*)d_ws;
    float* Qh = ws;
    float* Kh = Qh + (size_t)N * 128;
    float* Vh = Kh + (size_t)N * 128;
    float* wV = Vh + (size_t)N * 128;
    float* zb = wV + (size_t)N * 128;
    __bf16* pk = (__bf16*)(zb + (size_t)N * 8);     // N*2080 bytes in: 32B aligned
    __bf16* Qp = pk;                                 // 16384 bf16 each
    __bf16* Kp = pk + 16384;
    __bf16* Vp = pk + 2 * 16384;
    __bf16* Ep = pk + 3 * 16384;
    __bf16* Rp = pk + 4 * 16384;

    float* h_out = (float*)d_out;                 // [N, H, D] flat
    float* e_out = h_out + (size_t)N * 128;       // [E, H, D] flat

    pack_weights_kernel<<<(5 * 16384 + 255) / 256, 256, 0, stream>>>(Qw, Kw, Vw, Ew, Rw, pk);

    const size_t zcount = (size_t)N * 128 + (size_t)N * 8;  // wV and z adjacent
    zero_kernel<<<(unsigned)((zcount + 255) / 256), 256, 0, stream>>>(wV, zcount);

    // ---- node projections: guard-free kernel on fully-valid blocks ----
    const int nfull = N / 64;                     // blocks with 64 valid rows
    if (nfull > 0)
        node_proj_kernel<true><<<nfull, 128, 0, stream>>>(h, Qp, Qb, Kp, Kb, Vp, Vb,
                                                          Qh, Kh, Vh, 0, N);
    if (N - nfull * 64 > 0)                       // at most one remainder block
        node_proj_kernel<false><<<1, 128, 0, stream>>>(h, Qp, Qb, Kp, Kb, Vp, Vb,
                                                       Qh, Kh, Vh, nfull, N);

    // ---- fused edge kernel ----
    const int efull = E / 64;
    if (efull > 0)
        edge_kernel<true><<<efull, 128, 0, stream>>>(e, kr, src, dst, Ep, Eb, Rp, Rb,
                                                     Qh, Kh, Vh, e_out, wV, zb, 0, E);
    if (E - efull * 64 > 0)
        edge_kernel<false><<<1, 128, 0, stream>>>(e, kr, src, dst, Ep, Eb, Rp, Rb,
                                                  Qh, Kh, Vh, e_out, wV, zb, efull, E);

    const size_t hcount = (size_t)N * 128;
    finalize_kernel<<<(unsigned)((hcount + 255) / 256), 256, 0, stream>>>(wV, zb, h_out, N);
}
